// RBF_GAT_78151224918209
// MI455X (gfx1250) — matrix operand order repaired and checked by reading, not compiled, not run on hardware
//
#include <hip/hip_runtime.h>

typedef __attribute__((ext_vector_type(16))) _Float16 v16h;
typedef __attribute__((ext_vector_type(8)))  _Float16 v8h;
typedef __attribute__((ext_vector_type(8)))  float    v8f;

namespace {
constexpr int N    = 50000;     // nodes
constexpr int E    = 400000;    // edges
constexpr int H    = 8;         // heads
constexpr int D    = 128;       // D_IN == H*D_OUT
constexpr int NT_N = N / 16;    // 3125 node row-tiles
constexpr int NT_E = E / 16;    // 25000 edge row-tiles
constexpr int FRAG_HALVES = 128 * 128;   // halves per weight matrix fragment set (32 KB)
}

// order-preserving float <-> uint encoding for atomicMax-based segment max
__device__ __forceinline__ unsigned fenc(float f) {
  unsigned u = __float_as_uint(f);
  return (u & 0x80000000u) ? ~u : (u | 0x80000000u);
}
__device__ __forceinline__ float fdec(unsigned u) {
  return (u & 0x80000000u) ? __uint_as_float(u & 0x7FFFFFFFu)
                           : __uint_as_float(~u);
}

// ---- init: h_out = 0, denom = 0, emax = enc(-inf) ------------------------
__global__ void __launch_bounds__(256) k_init(float* h_out, float* denom,
                                              unsigned* emax) {
  int tid = blockIdx.x * 256 + threadIdx.x;          // N*D threads
  h_out[tid] = 0.0f;
  if (tid < N * H) { denom[tid] = 0.0f; emax[tid] = 0x007FFFFFu; }  // enc(-inf)
}

// ---- pre-swizzle the four 128x128 f32 weights into f16 WMMA B-fragments --
// global layout: [mat][ct(8)][kt(4)][plane(2)][lane(32)][8 halves]
//   plane p holds v16h elements p*8 .. p*8+7 (16B per lane -> conflict-free ds_load_b128)
// B 32x16 f16 layout: lane = 16*half + n holds column n; elems 0..7 = K 8*half+0..7, elems 8..15 = K 16+8*half+0..7
__global__ void __launch_bounds__(256) k_prep(const float* Wni, const float* Wnj,
                                              const float* Wnode, const float* Wfij,
                                              _Float16* bfrag) {
  int idx  = blockIdx.x * 256 + threadIdx.x;   // [0, 4*128*128)
  int ep   = idx & 7;
  int lane = (idx >> 3) & 31;
  int p    = (idx >> 8) & 1;
  int kt   = (idx >> 9) & 3;
  int ct   = (idx >> 11) & 7;
  int mat  = idx >> 14;
  int n = lane & 15, half = lane >> 4;
  int e   = p * 8 + ep;
  int K   = kt * 32 + 16 * p + 8 * half + ep;
  int col = ct * 16 + n;
  const float* W = (mat == 0) ? Wni : (mat == 1) ? Wnj : (mat == 2) ? Wnode : Wfij;
  bfrag[idx] = (_Float16)W[K * D + col];
}

// ---- async global->LDS staging of one 32KB fragment set (issue only) -----
__device__ __forceinline__ void stage_issue(const _Float16* gsrc, const _Float16* lds,
                                            int tid) {
  // generic LDS pointer low 32 bits == LDS byte offset (flat aperture encoding)
  unsigned lbase = (unsigned)(uintptr_t)lds;
  uint64_t gbase = (uint64_t)(uintptr_t)gsrc;
  #pragma unroll
  for (int i = 0; i < 8; ++i) {                 // 256 thr * 8 * 16B = 32 KB
    unsigned loff = lbase + (unsigned)(tid * 16 + i * 4096);
    uint64_t ga   = gbase + (uint64_t)(tid * 16 + i * 4096);
    asm volatile("global_load_async_to_lds_b128 %0, %1, off"
                 :: "v"(loff), "v"(ga) : "memory");
  }
}
__device__ __forceinline__ void stage_wait() {
  asm volatile("s_wait_asynccnt 0x0" ::: "memory");
}

// ---- A-fragment load: 16x32 f16 tile, lane = M, elems 0..7 = K 8*half+0..7,
//      elems 8..15 = K 16+8*half+0..7 (per ISA table); 4 k-tiles cover K=128 --
__device__ __forceinline__ void load_a_frags(const float* rowp, int half, v16h a[4]) {
  #pragma unroll
  for (int kt = 0; kt < 4; ++kt) {
    const float4* p = reinterpret_cast<const float4*>(rowp + kt * 32 + 8 * half);
    const float4* q = reinterpret_cast<const float4*>(rowp + kt * 32 + 16 + 8 * half);
    float4 x0 = p[0], x1 = p[1], x2 = q[0], x3 = q[1];
    v16h t = { (_Float16)x0.x, (_Float16)x0.y, (_Float16)x0.z, (_Float16)x0.w,
               (_Float16)x1.x, (_Float16)x1.y, (_Float16)x1.z, (_Float16)x1.w,
               (_Float16)x2.x, (_Float16)x2.y, (_Float16)x2.z, (_Float16)x2.w,
               (_Float16)x3.x, (_Float16)x3.y, (_Float16)x3.z, (_Float16)x3.w };
    a[kt] = t;
  }
}

// read one B fragment (ct,kt) for this lane from staged LDS
__device__ __forceinline__ v16h load_b_lds(const _Float16* sB, int f, int lane) {
  const v8h* bl = reinterpret_cast<const v8h*>(sB);
  v8h lo = bl[f * 64 + lane];        // plane 0: +0B,   lane stride 16B
  v8h hi = bl[f * 64 + 32 + lane];   // plane 1: +512B
  return __builtin_shufflevector(lo, hi, 0, 1, 2, 3, 4, 5, 6, 7,
                                 8, 9, 10, 11, 12, 13, 14, 15);
}

// ---- node kernel: one wave = one 16-row tile; 3 GEMMs, weights staged in LDS
__global__ void __launch_bounds__(256) k_node(const float* nfeats, const _Float16* bfrag,
                                              const float* b_node,
                                              float* f_ni, float* f_nj, float* hproj) {
  __shared__ alignas(16) _Float16 sB[FRAG_HALVES];   // 32 KB, reused per matrix
  int tid  = threadIdx.x;
  int wave = tid >> 5, lane = tid & 31;
  int tile = blockIdx.x * 8 + wave;
  bool active = tile < NT_N;
  int tile_c = active ? tile : NT_N - 1;             // clamp: keep EXEC full for WMMA
  int n = lane & 15, half = lane >> 4;

  stage_issue(bfrag, sB, tid);                       // mat 0 overlapped with A loads
  v16h a[4];
  load_a_frags(nfeats + (size_t)(tile_c * 16 + n) * D, half, a);
  stage_wait();
  __syncthreads();

  float* outs[3] = { f_ni, f_nj, hproj };
  for (int mat = 0; mat < 3; ++mat) {
    float* out = outs[mat];
    for (int ct = 0; ct < 8; ++ct) {
      v16h b0 = load_b_lds(sB, ct * 4 + 0, lane);
      v16h b1 = load_b_lds(sB, ct * 4 + 1, lane);
      v16h b2 = load_b_lds(sB, ct * 4 + 2, lane);
      v16h b3 = load_b_lds(sB, ct * 4 + 3, lane);
      v8f c = {};
      c = __builtin_amdgcn_wmma_f32_16x16x32_f16(false, a[0], false, b0, (short)0, c, false, false);
      c = __builtin_amdgcn_wmma_f32_16x16x32_f16(false, a[1], false, b1, (short)0, c, false, false);
      c = __builtin_amdgcn_wmma_f32_16x16x32_f16(false, a[2], false, b2, (short)0, c, false, false);
      c = __builtin_amdgcn_wmma_f32_16x16x32_f16(false, a[3], false, b3, (short)0, c, false, false);
      int col = ct * 16 + n;
      if (active) {
        float badd = (mat == 2) ? b_node[col] : 0.0f;
        #pragma unroll
        for (int v = 0; v < 8; ++v) {               // C: lane=N, VGPR v -> M = v + 8*half
          int row = tile * 16 + v + 8 * half;
          out[(size_t)row * D + col] = (mat == 2) ? (c[v] + badd) : c[v];
        }
      }
    }
    if (mat < 2) {                                   // restage next weight matrix
      __syncthreads();
      stage_issue(bfrag + (size_t)(mat + 1) * FRAG_HALVES, sB, tid);
      stage_wait();
      __syncthreads();
    }
  }
}

// ---- edge kernel: GEMM(efeats, W_fij) fused with gather-add, leaky-relu,
//      f_out store, per-head attn dot product and atomic segment-max --------
__global__ void __launch_bounds__(256) k_edge(const float* efeats, const _Float16* bfrag,
                                              const int* src, const int* dst,
                                              const float* f_ni, const float* f_nj,
                                              const float* bias, const float* attn,
                                              float* f_out, float* evals, unsigned* emax) {
  __shared__ alignas(16) _Float16 sB[FRAG_HALVES];   // 32 KB: W_fij fragments
  int tid  = threadIdx.x;
  int wave = tid >> 5, lane = tid & 31;
  int tile = blockIdx.x * 8 + wave;                  // NT_E divisible by 8: all active
  int n = lane & 15, half = lane >> 4;

  stage_issue(bfrag + (size_t)3 * FRAG_HALVES, sB, tid);
  v16h a[4];
  load_a_frags(efeats + (size_t)(tile * 16 + n) * D, half, a);
  stage_wait();
  __syncthreads();

  for (int ct = 0; ct < 8; ++ct) {                   // one 16-col tile == one head
    v16h b0 = load_b_lds(sB, ct * 4 + 0, lane);
    v16h b1 = load_b_lds(sB, ct * 4 + 1, lane);
    v16h b2 = load_b_lds(sB, ct * 4 + 2, lane);
    v16h b3 = load_b_lds(sB, ct * 4 + 3, lane);
    v8f c = {};
    c = __builtin_amdgcn_wmma_f32_16x16x32_f16(false, a[0], false, b0, (short)0, c, false, false);
    c = __builtin_amdgcn_wmma_f32_16x16x32_f16(false, a[1], false, b1, (short)0, c, false, false);
    c = __builtin_amdgcn_wmma_f32_16x16x32_f16(false, a[2], false, b2, (short)0, c, false, false);
    c = __builtin_amdgcn_wmma_f32_16x16x32_f16(false, a[3], false, b3, (short)0, c, false, false);
    int col = ct * 16 + n;
    float av = attn[col];                           // attn[H][16] row-major
    float bs = bias[col];
    #pragma unroll
    for (int v = 0; v < 8; ++v) {
      int erow = tile * 16 + v + 8 * half;
      int s = src[erow], d = dst[erow];
      float x = c[v] + f_ni[(size_t)s * D + col] + f_nj[(size_t)d * D + col] + bs;
      float fo = x > 0.0f ? x : 0.01f * x;          // leaky_relu(0.01)
      __builtin_nontemporal_store(fo, &f_out[(size_t)erow * D + col]);
      float t = fo * av;                            // reduce over 16 lanes (D_OUT dim)
      t += __shfl_xor(t, 1, 16);
      t += __shfl_xor(t, 2, 16);
      t += __shfl_xor(t, 4, 16);
      t += __shfl_xor(t, 8, 16);
      if (n == 0) {
        evals[erow * H + ct] = t;
        atomicMax(&emax[d * H + ct], fenc(t));
      }
    }
  }
}

// ---- softmax pass 1: ex = exp(e - emax[dst]); denom[dst] += ex ------------
__global__ void __launch_bounds__(256) k_soft(const int* dst, const unsigned* emax,
                                              float* evals, float* denom) {
  int tid = blockIdx.x * 256 + threadIdx.x;         // E*H threads
  int edge = tid >> 3, hd = tid & 7;
  int d = dst[edge];
  float m  = fdec(emax[d * H + hd]);
  float ex = __expf(evals[tid] - m);
  evals[tid] = ex;
  atomicAdd(&denom[d * H + hd], ex);
}

// ---- scatter: h_out[dst] += h[src] * (ex/denom[dst]) ----------------------
__global__ void __launch_bounds__(256) k_scatter(const int* src, const int* dst,
                                                 const float* hproj, const float* evals,
                                                 const float* denom, float* h_out) {
  int tid  = blockIdx.x * 256 + threadIdx.x;        // E*D threads
  int edge = tid >> 7, col = tid & 127, hd = col >> 4;
  int s = src[edge], d = dst[edge];
  float a = evals[edge * H + hd] / denom[d * H + hd];
  atomicAdd(&h_out[(size_t)d * D + col], hproj[(size_t)s * D + col] * a);
}

extern "C" void kernel_launch(void* const* d_in, const int* in_sizes, int n_in,
                              void* d_out, int out_size, void* d_ws, size_t ws_size,
                              hipStream_t stream) {
  const float* nfeats = (const float*)d_in[0];
  const float* efeats = (const float*)d_in[1];
  const int*   src    = (const int*)d_in[2];
  const int*   dst    = (const int*)d_in[3];
  const float* W_node = (const float*)d_in[4];
  const float* b_node = (const float*)d_in[5];
  const float* W_ni   = (const float*)d_in[6];
  const float* W_nj   = (const float*)d_in[7];
  const float* W_fij  = (const float*)d_in[8];
  const float* attn   = (const float*)d_in[9];
  const float* bias   = (const float*)d_in[10];

  float* h_out = (float*)d_out;                       // [N,H,16]
  float* f_out = h_out + (size_t)N * D;               // [E,H,16]

  // workspace layout (~93 MB total)
  float*    f_ni  = (float*)d_ws;                     // N*D f32
  float*    f_nj  = f_ni  + (size_t)N * D;            // N*D f32
  float*    hproj = f_nj  + (size_t)N * D;            // N*D f32
  float*    evals = hproj + (size_t)N * D;            // E*H f32 (e, then ex in place)
  float*    denom = evals + (size_t)E * H;            // N*H f32
  unsigned* emax  = (unsigned*)(denom + (size_t)N * H);       // N*H u32 (encoded)
  _Float16* bfrag = (_Float16*)(emax + (size_t)N * H);        // 4*128*128 f16

  k_init   <<<(N * D) / 256,     256, 0, stream>>>(h_out, denom, emax);
  k_prep   <<<(4 * D * D) / 256, 256, 0, stream>>>(W_ni, W_nj, W_node, W_fij, bfrag);
  k_node   <<<(NT_N + 7) / 8,    256, 0, stream>>>(nfeats, bfrag, b_node,
                                                   f_ni, f_nj, hproj);
  k_edge   <<<NT_E / 8,          256, 0, stream>>>(efeats, bfrag, src, dst,
                                                   f_ni, f_nj, bias, attn,
                                                   f_out, evals, emax);
  k_soft   <<<(E * H) / 256,     256, 0, stream>>>(dst, emax, evals, denom);
  k_scatter<<<(E * D) / 256,     256, 0, stream>>>(src, dst, hproj, evals, denom, h_out);
}
